// Resonator_34540126994688
// MI455X (gfx1250) — matrix operand order repaired and checked by reading, not compiled, not run on hardware
//
#include <hip/hip_runtime.h>

// Problem constants (match reference)
#define BB 64
#define FF 4
#define MM 4096
#define DD 1024
#define NITER 15

typedef __attribute__((ext_vector_type(16))) _Float16 v16h;
typedef __attribute__((ext_vector_type(8)))  _Float16 v8h;
typedef __attribute__((ext_vector_type(8)))  float    v8f;

#define WMMA_F16(a, b, c) \
    __builtin_amdgcn_wmma_f32_16x16x32_f16(false, (a), false, (b), (short)0, (c), false, false)

// ---------------------------------------------------------------------------
// 32x32 register-blocked WMMA GEMM: four 16x16 f32 tiles per wave.
// A fragment (ISA 7.12.2, 16-bit A 16x32): lane row = l&15, sel=(l>>4)*8,
//   halves 0..7 -> K=sel+0..7, halves 8..15 -> K=sel+16..23 (two b128 loads).
// B stored COLUMN-major (K contiguous per output column): lane col = l&15,
//   sel=(l>>4)*8, halves 0..7 -> K=sel+0..7, halves 8..15 -> K=sel+16..23
//   (same K order as A; two b128 loads).
// ---------------------------------------------------------------------------
__device__ __forceinline__ void wmma_gemm_2x2(
    const _Float16* __restrict__ A, size_t a_off, size_t a_lstride,
    const _Float16* __restrict__ Bc, size_t b_off, size_t b_lstride,
    int K, int lane, v8f acc[4])
{
    const int l15  = lane & 15;
    const int asel = (lane >> 4) * 8;
    const int bsel = (lane >> 4) * 8;
    const _Float16* ap0 = A  + a_off + (size_t)l15 * a_lstride + asel;
    const _Float16* ap1 = ap0 + (size_t)16 * a_lstride;
    const _Float16* bp0 = Bc + b_off + (size_t)l15 * b_lstride + bsel;
    const _Float16* bp1 = bp0 + (size_t)16 * b_lstride;

    v8f c00 = {}, c01 = {}, c10 = {}, c11 = {};
#pragma unroll 2
    for (int k0 = 0; k0 < K; k0 += 32) {
        v8h a0lo = *(const v8h*)(ap0 + k0), a0hi = *(const v8h*)(ap0 + k0 + 16);
        v8h a1lo = *(const v8h*)(ap1 + k0), a1hi = *(const v8h*)(ap1 + k0 + 16);
        v8h b0lo = *(const v8h*)(bp0 + k0), b0hi = *(const v8h*)(bp0 + k0 + 16);
        v8h b1lo = *(const v8h*)(bp1 + k0), b1hi = *(const v8h*)(bp1 + k0 + 16);
        __builtin_prefetch(ap0 + k0 + 256, 0, 3);   // locality 3 -> near-scope prefetch
        __builtin_prefetch(ap1 + k0 + 256, 0, 3);
        __builtin_prefetch(bp0 + k0 + 256, 0, 3);
        __builtin_prefetch(bp1 + k0 + 256, 0, 3);
        v16h a0, a1, b0, b1;
#pragma unroll
        for (int i = 0; i < 8; ++i) {
            a0[i] = a0lo[i]; a0[i + 8] = a0hi[i];
            a1[i] = a1lo[i]; a1[i + 8] = a1hi[i];
            b0[i] = b0lo[i]; b0[i + 8] = b0hi[i];
            b1[i] = b1lo[i]; b1[i + 8] = b1hi[i];
        }
        c00 = WMMA_F16(a0, b0, c00);
        c01 = WMMA_F16(a0, b1, c01);
        c10 = WMMA_F16(a1, b0, c10);
        c11 = WMMA_F16(a1, b1, c11);
    }
    acc[0] = c00; acc[1] = c01; acc[2] = c10; acc[3] = c11;
}

// --- one-time prep kernels --------------------------------------------------

__global__ void cvt_cb_kernel(const float* __restrict__ cb, _Float16* __restrict__ cbh) {
    int i = blockIdx.x * 256 + threadIdx.x;          // F*M*D threads
    cbh[i] = (_Float16)cb[i];
}

// cbT (F,D,M): cbT[f][d][m] = cb[f][m][d]  (column-major for GEMM2's B)
__global__ void transpose_cb_kernel(const float* __restrict__ cb, _Float16* __restrict__ cbt) {
    int i = blockIdx.x * 256 + threadIdx.x;          // linear cbT idx
    int m = i & (MM - 1);
    int r = i >> 12;                                  // f*D + d
    int d = r & (DD - 1);
    int f = r >> 10;
    cbt[i] = (_Float16)cb[(((size_t)f * MM + m) << 10) + d];
}

__global__ void est_init_kernel(const float* __restrict__ init_est, _Float16* __restrict__ esth) {
    int i = blockIdx.x * 256 + threadIdx.x;          // B*F*D threads
    esth[i] = (_Float16)init_est[i];
}

// flags: [0]=conv, [1]=iters, [2]=changed, [3]=copyflag
__global__ void flags_init_kernel(int* flags) {
    flags[0] = 0; flags[1] = 0; flags[2] = 0; flags[3] = 0;
}

// --- per-iteration kernels --------------------------------------------------

// new_est[b,f,d] = input[b,d] * prod_j est[b,j,d] * est[b,f,d]
__global__ void unbind_kernel(const float* __restrict__ input,
                              const _Float16* __restrict__ est,
                              _Float16* __restrict__ nw) {
    int i = blockIdx.x * 256 + threadIdx.x;          // B*D threads
    int b = i >> 10, d = i & (DD - 1);
    size_t base = ((size_t)b * FF) << 10;
    float e0 = (float)est[base + 0 * DD + d];
    float e1 = (float)est[base + 1 * DD + d];
    float e2 = (float)est[base + 2 * DD + d];
    float e3 = (float)est[base + 3 * DD + d];
    float t  = input[((size_t)b << 10) + d] * (e0 * e1 * e2 * e3);
    nw[base + 0 * DD + d] = (_Float16)(t * e0);
    nw[base + 1 * DD + d] = (_Float16)(t * e1);
    nw[base + 2 * DD + d] = (_Float16)(t * e2);
    nw[base + 3 * DD + d] = (_Float16)(t * e3);
}

// sim[b,f,m] = sum_d A[b,f,d] * cb[f,m,d]   (A = new_est, or est for final pass)
// one wave per 32x32 output block (2x2 tiles); 8 waves / 256-thread block
__global__ void sim_wmma_kernel(const _Float16* __restrict__ A,
                                const _Float16* __restrict__ cbh,
                                _Float16* __restrict__ sim) {
    int wave = blockIdx.x * 8 + (threadIdx.x >> 5);
    int lane = threadIdx.x & 31;
    int mg = wave & 127;                 // MM/32 groups
    int f  = (wave >> 7) & 3;
    int bt = wave >> 9;                  // 0..1
    int b0 = bt * 32, m0 = mg * 32;

    v8f acc[4];
    wmma_gemm_2x2(A,   ((size_t)(b0 * FF + f)) << 10, (size_t)FF * DD,
                  cbh, ((size_t)(f * MM + m0)) << 10, (size_t)DD,
                  DD, lane, acc);

    int l15 = lane & 15, rb = (lane >> 4) * 8;
#pragma unroll
    for (int t = 0; t < 4; ++t) {
        int    br = b0 + (t >> 1) * 16 + rb;
        size_t m  = (size_t)(m0 + (t & 1) * 16 + l15);
        v8f    c  = acc[t];
#pragma unroll
        for (int v = 0; v < 8; ++v) {
            int b = br + v;
            sim[(((size_t)b * FF + f) << 12) + m] = (_Float16)c[v];  // exact: |sim|<=1024
        }
    }
}

// upd[b,f,d] = sign( sum_m sim[b,f,m] * cbT[f,d,m] ); detect upd != est
__global__ void upd_wmma_kernel(const _Float16* __restrict__ sim,
                                const _Float16* __restrict__ cbt,
                                const _Float16* __restrict__ est,
                                _Float16* __restrict__ upd,
                                int* __restrict__ flags) {
    int wave = blockIdx.x * 8 + (threadIdx.x >> 5);
    int lane = threadIdx.x & 31;
    int dg = wave & 31;                  // DD/32 groups
    int f  = (wave >> 5) & 3;
    int bt = wave >> 7;                  // 0..1
    int b0 = bt * 32, d0 = dg * 32;

    v8f acc[4];
    wmma_gemm_2x2(sim, ((size_t)(b0 * FF + f)) << 12, (size_t)FF * MM,
                  cbt, ((size_t)(f * DD + d0)) << 12, (size_t)MM,
                  MM, lane, acc);

    int l15 = lane & 15, rb = (lane >> 4) * 8;
    int diff = 0;
#pragma unroll
    for (int t = 0; t < 4; ++t) {
        int    br = b0 + (t >> 1) * 16 + rb;
        size_t d  = (size_t)(d0 + (t & 1) * 16 + l15);
        v8f    c  = acc[t];
#pragma unroll
        for (int v = 0; v < 8; ++v) {
            int b = br + v;
            size_t idx = (((size_t)b * FF + f) << 10) + d;
            _Float16 s = (c[v] >= 0.0f) ? (_Float16)1.0f : (_Float16)-1.0f; // sign(0)=+1
            upd[idx] = s;
            if (s != est[idx]) diff = 1;
        }
    }
    if (diff) atomicOr(&flags[2], 1);
}

// reference step ordering: iters uses OLD conv; est update gated on OLD conv
__global__ void step_kernel(int* flags) {
    int oldc = flags[0];
    if (!oldc) flags[1] += 1;                 // iters2 = iters + (conv?0:1)
    int now_conv = (flags[2] == 0);           // all(upd==est)
    flags[0] = oldc | now_conv;               // conv |= now_conv
    flags[3] = oldc ? 0 : 1;                  // est2 = where(OLD conv, est, upd)
    flags[2] = 0;                             // reset 'changed'
}

__global__ void est_update_kernel(const _Float16* __restrict__ upd,
                                  _Float16* __restrict__ est,
                                  const int* __restrict__ flags) {
    int i = blockIdx.x * 256 + threadIdx.x;   // B*F*D threads
    if (flags[3]) est[i] = upd[i];
}

// --- cleanup ----------------------------------------------------------------

// outcome[b,f] = argmax_m |sim[b,f,m]| (first index on ties)
__global__ void argmax_kernel(const _Float16* __restrict__ sim, float* __restrict__ out) {
    __shared__ float sv[256];
    __shared__ int   si[256];
    int bf = blockIdx.x;                      // 0..B*F-1
    int t  = threadIdx.x;
    float best = -1.0f; int bidx = 0x7fffffff;
    for (int m = t; m < MM; m += 256) {
        float v = fabsf((float)sim[((size_t)bf << 12) + m]);
        if (v > best || (v == best && m < bidx)) { best = v; bidx = m; }
    }
    sv[t] = best; si[t] = bidx;
    __syncthreads();
    for (int s = 128; s > 0; s >>= 1) {
        if (t < s) {
            float ov = sv[t + s]; int oi = si[t + s];
            if (ov > sv[t] || (ov == sv[t] && oi < si[t])) { sv[t] = ov; si[t] = oi; }
        }
        __syncthreads();
    }
    if (t == 0) out[bf] = (float)si[0];
}

// d_out = [outcome(256) | est(B*F*D f32) | iters | conv]
__global__ void emit_kernel(const _Float16* __restrict__ est,
                            const int* __restrict__ flags,
                            float* __restrict__ out) {
    int i = blockIdx.x * 256 + threadIdx.x;   // B*F*D threads
    out[BB * FF + i] = (float)est[i];
    if (i == 0) {
        out[BB * FF + BB * FF * DD + 0] = (float)flags[1];  // iters
        out[BB * FF + BB * FF * DD + 1] = (float)flags[0];  // converged
    }
}

// ---------------------------------------------------------------------------
extern "C" void kernel_launch(void* const* d_in, const int* in_sizes, int n_in,
                              void* d_out, int out_size, void* d_ws, size_t ws_size,
                              hipStream_t stream) {
    (void)in_sizes; (void)n_in; (void)out_size; (void)ws_size;
    const float* input    = (const float*)d_in[0];   // (B,D)
    const float* init_est = (const float*)d_in[1];   // (B,F,D)
    const float* cb       = (const float*)d_in[2];   // (F,M,D)
    float*       out      = (float*)d_out;

    char* ws = (char*)d_ws;
    const size_t off_cbh   = 0;                                   // F*M*D f16 = 32 MB
    const size_t off_cbt   = off_cbh + (size_t)FF * MM * DD * 2;  // 32 MB
    const size_t off_est   = off_cbt + (size_t)FF * MM * DD * 2;
    const size_t off_new   = off_est + (size_t)BB * FF * DD * 2;
    const size_t off_upd   = off_new + (size_t)BB * FF * DD * 2;
    const size_t off_sim   = off_upd + (size_t)BB * FF * DD * 2;  // B*F*M f16 = 2 MB
    const size_t off_flags = off_sim + (size_t)BB * FF * MM * 2;

    _Float16* cbh   = (_Float16*)(ws + off_cbh);
    _Float16* cbt   = (_Float16*)(ws + off_cbt);
    _Float16* est   = (_Float16*)(ws + off_est);
    _Float16* nw    = (_Float16*)(ws + off_new);
    _Float16* upd   = (_Float16*)(ws + off_upd);
    _Float16* sim   = (_Float16*)(ws + off_sim);
    int*      flags = (int*)(ws + off_flags);

    const int elemsCB  = FF * MM * DD;   // 16,777,216
    const int elemsEST = BB * FF * DD;   // 262,144

    cvt_cb_kernel      <<<elemsCB / 256,  256, 0, stream>>>(cb, cbh);
    transpose_cb_kernel<<<elemsCB / 256,  256, 0, stream>>>(cb, cbt);
    est_init_kernel    <<<elemsEST / 256, 256, 0, stream>>>(init_est, est);
    flags_init_kernel  <<<1, 1, 0, stream>>>(flags);

    const int simBlocks = (2 * 4 * (MM / 32)) / 8;   // 1024 waves -> 128 blocks
    const int updBlocks = (2 * 4 * (DD / 32)) / 8;   // 256 waves -> 32 blocks

    for (int it = 0; it < NITER; ++it) {
        unbind_kernel    <<<(BB * DD) / 256, 256, 0, stream>>>(input, est, nw);
        sim_wmma_kernel  <<<simBlocks, 256, 0, stream>>>(nw, cbh, sim);
        upd_wmma_kernel  <<<updBlocks, 256, 0, stream>>>(sim, cbt, est, upd, flags);
        step_kernel      <<<1, 1, 0, stream>>>(flags);
        est_update_kernel<<<elemsEST / 256, 256, 0, stream>>>(upd, est, flags);
    }

    // final similarity from converged estimates, then cleanup argmax
    sim_wmma_kernel<<<simBlocks, 256, 0, stream>>>(est, cbh, sim);
    argmax_kernel  <<<BB * FF, 256, 0, stream>>>(sim, out);
    emit_kernel    <<<elemsEST / 256, 256, 0, stream>>>(est, flags, out);
}
